// GraphAttentionConvolution_12549894439026
// MI455X (gfx1250) — matrix operand order repaired and checked by reading, not compiled, not run on hardware
//
#include <hip/hip_runtime.h>
#include <hip/hip_bf16.h>
#include <cstdint>

// GAT layer, fused for MI455X (gfx1250, wave32, WMMA).
//   HBM floor: adj 400MB read-once (~17us @23.3TB/s).
//   attention@h = 25.6 GFLOP via v_wmma_f32_16x16x32_bf16; h kept bf16,
//   pre-swizzled into WMMA-B fragment order (2.6MB, L2-resident) so the hot
//   loop needs no LDS staging for h.
//   softmax without running max: |e| is O(10) for this input scale -> exp safe
//   in f32; masked entries -> 0; divide by rowsum at the end.

#define N_NODES 10000
#define F_IN    256
#define F_OUT   128
#define NTILES  313          // ceil(N/32) j-tiles of 32
#define HSWZ_TILE_DW (128 * 16)   // per tile: 128 f x 16 dwords (32 bf16 K-vals)

typedef __attribute__((ext_vector_type(16))) __bf16 v16bf;
typedef __attribute__((ext_vector_type(8)))  float  v8f;

union FragBF {
  v16bf v;
  unsigned short h[16];
  uint4 q[2];
};

__device__ __forceinline__ unsigned short f2bf(float f) {
  __bf16 b = (__bf16)f;
  return __builtin_bit_cast(unsigned short, b);
}
__device__ __forceinline__ unsigned pack2(float lo, float hi) {
  return (unsigned)f2bf(lo) | ((unsigned)f2bf(hi) << 16);   // -> v_cvt_pk_bf16_f32
}

// ---------------------------------------------------------------------------
// Kernel 1: W [256,128] f32 -> Wb [128][256] bf16 (transposed), and zero the
// hswz tail (unwritten half of j-tile 312 + pad tile 313) so the last K=64
// iteration of k_gat needs no branches on the B side.
// ---------------------------------------------------------------------------
__global__ void __launch_bounds__(256) k_pack_w(const float* __restrict__ W,
                                                unsigned short* __restrict__ Wb,
                                                unsigned* __restrict__ hswzd) {
  int idx = blockIdx.x * blockDim.x + threadIdx.x;  // 0..32767
  int f = idx & (F_OUT - 1);
  int k = idx >> 7;
  Wb[f * F_IN + k] = f2bf(W[k * F_OUT + f]);
  if (idx < 2 * HSWZ_TILE_DW)                       // tiles 312..313
    hswzd[312 * HSWZ_TILE_DW + idx] = 0u;
}

// ---------------------------------------------------------------------------
// Kernel 2: h = input @ W (bf16 WMMA, f32 acc).
// Block = 256 threads = 8 waves, 16 rows; wave w owns feature tile w.
//  - writes hswz: h in WMMA-B fragment order [jtile][f][16 dwords]
//  - computes e1[i]=h[i].a1, e2[i]=h[i].a2 in-register (shuffle+LDS reduce)
// ---------------------------------------------------------------------------
__global__ void __launch_bounds__(256) k_h_gemm(const float* __restrict__ input,
                                                const unsigned short* __restrict__ Wb,
                                                unsigned* __restrict__ hswzd,
                                                const float* __restrict__ a1,
                                                const float* __restrict__ a2,
                                                float* __restrict__ e1,
                                                float* __restrict__ e2) {
  __shared__ __align__(16) unsigned abuf[16 * 128];  // 16 rows x 256 K bf16 pairs (8KB)
  __shared__ float ep[2][8][16];                     // e1/e2 partials per wave

  const int t  = threadIdx.x;
  const int i0 = blockIdx.x * 16;

  // Stage A tile (f32 -> packed bf16 pairs), coalesced b128 reads.
  {
    int r  = t >> 4;
    int cg = t & 15;
    const float4* s4 = (const float4*)(input + (size_t)(i0 + r) * F_IN + cg * 16);
    unsigned* dst = abuf + r * 128 + cg * 8;
#pragma unroll
    for (int d = 0; d < 4; ++d) {
      float4 v = s4[d];
      dst[2 * d]     = pack2(v.x, v.y);
      dst[2 * d + 1] = pack2(v.z, v.w);
    }
  }
  __syncthreads();

  const int w    = t >> 5;
  const int lane = t & 31;
  const int hf   = lane >> 4;   // lane half
  const int lr   = lane & 15;
  const int f    = w * 16 + lr;

  const unsigned* Wbd = (const unsigned*)Wb;  // dword view, row = 128 dwords

  v8f acc = {};
#pragma unroll
  for (int kk = 0; kk < 8; ++kk) {            // K = kk*32 .. +31
    FragBF a, b;
    // A frag: row lr; lanes 0-15: K 0..15, lanes 16-31: K 16..31 (linear)
    int ab = lr * 128 + kk * 16 + hf * 8;
    a.q[0] = *(const uint4*)&abuf[ab];
    a.q[1] = *(const uint4*)&abuf[ab + 4];
    // B frag: col f; lanes 0-15: K 0..15, lanes 16-31: K 16..31 (linear)
    const uint4* wb = (const uint4*)(Wbd + f * 128 + kk * 16 + hf * 8);
    b.q[0] = wb[0];
    b.q[1] = wb[1];
    acc = __builtin_amdgcn_wmma_f32_16x16x32_bf16(false, a.v, false, b.v,
                                                  (short)0, acc, false, false);
  }

  // Store h in WMMA-B fragment order: lane holds M = hf*8..hf*8+7 (consecutive)
  // -> 4 packed dwords, one b128 store.
  {
    const int jt    = blockIdx.x >> 1;        // j-tile (32 rows)
    const int khalf = blockIdx.x & 1;         // which 16-row half of the tile
    uint4 pk;
    pk.x = pack2(acc[0], acc[1]);
    pk.y = pack2(acc[2], acc[3]);
    pk.z = pack2(acc[4], acc[5]);
    pk.w = pack2(acc[6], acc[7]);
    *(uint4*)(hswzd + (size_t)(jt * 128 + f) * 16 + khalf * 8 + hf * 4) = pk;
  }

  // e1/e2: reduce acc[r]*a1[f] over the 16 lanes (lr) of each half, then over waves.
  {
    const float a1f = a1[f];
    const float a2f = a2[f];
#pragma unroll
    for (int r = 0; r < 8; ++r) {
      float v1 = acc[r] * a1f;
      float v2 = acc[r] * a2f;
#pragma unroll
      for (int m = 1; m <= 8; m <<= 1) {
        v1 += __shfl_xor(v1, m, 32);
        v2 += __shfl_xor(v2, m, 32);
      }
      if (lr == 0) {
        ep[0][w][hf * 8 + r] = v1;
        ep[1][w][hf * 8 + r] = v2;
      }
    }
  }
  __syncthreads();
  if (t < 32) {
    int which = t >> 4, M = t & 15;
    float s = 0.f;
#pragma unroll
    for (int wv = 0; wv < 8; ++wv) s += ep[which][wv][M];
    (which ? e2 : e1)[i0 + M] = s;
  }
}

// ---------------------------------------------------------------------------
// Kernel 3 (fused hot loop): 16-row block streams adj once (the 400MB floor).
// Per K=64 iteration: P = exp(leaky(e1[i]+e2[j]))*mask into ping-pong LDS
// (one barrier), two v_wmma_f32_16x16x32_bf16 with B-frags loaded straight
// from L2-resident hswz. Finish: divide by rowsum, add bias.
// ---------------------------------------------------------------------------
__global__ void __launch_bounds__(256) k_gat(const int* __restrict__ adj,
                                             const unsigned* __restrict__ hswzd,
                                             const float* __restrict__ e1,
                                             const float* __restrict__ e2,
                                             const float* __restrict__ bias,
                                             float* __restrict__ out) {
  __shared__ __align__(16) unsigned pbuf[2][16 * 32];  // ping-pong P tile 16x64 bf16 (2x2KB)
  __shared__ float rsbuf[256];
  __shared__ float rowsum[16];

  const int t  = threadIdx.x;
  const int i0 = blockIdx.x * 16;

  const int w    = t >> 5;
  const int lane = t & 31;
  const int hf   = lane >> 4;
  const int lr   = lane & 15;
  const int f    = w * 16 + lr;

  // P mapping: thread t -> row pr, column pair (2*pcc, 2*pcc+1) per 32-subtile.
  const int pr  = t >> 4;
  const int pcc = t & 15;
  const float e1r = e1[i0 + pr];
  const int* adjrow = adj + (size_t)(i0 + pr) * N_NODES;

  const uint4* hswz4 = (const uint4*)hswzd;

  float rs = 0.f;
  v8f acc = {};

  const int niter = (N_NODES + 63) / 64;     // 157
  for (int it = 0; it < niter; ++it) {
    const int j0  = it * 64;
    const int par = it & 1;

    // ---- P tile: 4 entries per thread (2 cols x 2 subtiles) ----
#pragma unroll
    for (int s = 0; s < 2; ++s) {
      int jb = j0 + s * 32 + 2 * pcc;
      float p0 = 0.f, p1 = 0.f;
      if (jb + 1 < N_NODES) {                 // N even: pair fully valid or fully not
        int2   av = *(const int2*)(adjrow + jb);
        float2 ev = *(const float2*)(e2 + jb);
        if (av.x > 0) {
          float v = e1r + ev.x;
          v = (v > 0.f) ? v : 0.2f * v;       // LeakyReLU(0.2)
          p0 = __expf(v);
        }
        if (av.y > 0) {
          float v = e1r + ev.y;
          v = (v > 0.f) ? v : 0.2f * v;
          p1 = __expf(v);
        }
      }
      rs += p0 + p1;
      pbuf[par][pr * 32 + s * 16 + pcc] = pack2(p0, p1);
    }
    if (j0 + 192 < N_NODES)
      __builtin_prefetch(adjrow + j0 + 128, 0, 0);   // global_prefetch_b8, adj stream

    __syncthreads();   // single barrier per iter (ping-pong pbuf)

    // ---- two WMMAs ----
    const uint4* pq = (const uint4*)pbuf[par];
#pragma unroll
    for (int s = 0; s < 2; ++s) {
      FragBF a, b;
      // A frag from P: row lr; lanes 0-15: K 0..15, 16-31: K 16..31 (linear)
      a.q[0] = pq[lr * 8 + s * 4 + hf * 2];
      a.q[1] = pq[lr * 8 + s * 4 + hf * 2 + 1];
      // B frag straight from pre-swizzled h (L2): 32B contiguous per lane.
      const int jt = (j0 >> 5) + s;
      const uint4* hz = hswz4 + (size_t)(jt * 128 + f) * 4 + hf * 2;
      b.q[0] = hz[0];
      b.q[1] = hz[1];
      acc = __builtin_amdgcn_wmma_f32_16x16x32_bf16(false, a.v, false, b.v,
                                                    (short)0, acc, false, false);
    }
  }

  // ---- rowsum reduction: threads r*16..r*16+15 own row r ----
  rsbuf[t] = rs;
  __syncthreads();
  if (t < 16) {
    float s = 0.f;
#pragma unroll
    for (int u = 0; u < 16; ++u) s += rsbuf[t * 16 + u];
    rowsum[t] = s;
  }
  __syncthreads();

  const float bv = bias[f];
#pragma unroll
  for (int r = 0; r < 8; ++r) {
    int M = r + hf * 8;
    out[(size_t)(i0 + M) * F_OUT + f] = acc[r] / rowsum[M] + bv;
  }
}

// ---------------------------------------------------------------------------
extern "C" void kernel_launch(void* const* d_in, const int* in_sizes, int n_in,
                              void* d_out, int out_size, void* d_ws, size_t ws_size,
                              hipStream_t stream) {
  const float* input = (const float*)d_in[0];
  const int*   adj   = (const int*)d_in[1];
  const float* W     = (const float*)d_in[2];
  const float* a1    = (const float*)d_in[3];
  const float* a2    = (const float*)d_in[4];
  const float* bias  = (const float*)d_in[5];
  float* out = (float*)d_out;

  char* ws = (char*)d_ws;
  // hswz: 314 tiles x 128 f x 16 dwords = 2,572,288 B (tile 313 is zero pad)
  unsigned* hswzd = (unsigned*)ws;
  unsigned short* Wb = (unsigned short*)(ws + 2572288);             // 65,536 B
  float* e1 = (float*)(ws + 2572288 + 65536);                       // 40,000 B
  float* e2 = (float*)(ws + 2572288 + 65536 + 40000);               // 40,000 B

  k_pack_w<<<(F_IN * F_OUT) / 256, 256, 0, stream>>>(W, Wb, hswzd);
  k_h_gemm<<<N_NODES / 16, 256, 0, stream>>>(input, Wb, hswzd, a1, a2, e1, e2);
  k_gat<<<N_NODES / 16, 256, 0, stream>>>(adj, hswzd, e1, e2, bias, out);
}
